// ContextQueryAttention_85959475462395
// MI455X (gfx1250) — matrix operand order repaired and checked by reading, not compiled, not run on hardware
//
#include <hip/hip_runtime.h>
#include <stdint.h>

#define DEVINLINE __device__ __forceinline__

typedef __attribute__((ext_vector_type(16))) __bf16 v16bf;
typedef __attribute__((ext_vector_type(8)))  __bf16 v8bf;
typedef __attribute__((ext_vector_type(8)))  float  v8f;

constexpr int NB = 64;    // batch
constexpr int LQ = 256;   // query length
constexpr int LC = 2048;  // context length
constexpr int DD = 128;   // feature dim

// ---------------------------------------------------------------- WMMA helpers
DEVINLINE v8f wmma_bf16(v16bf a, v16bf b, v8f c) {
  // v_wmma_f32_16x16x32_bf16
  return __builtin_amdgcn_wmma_f32_16x16x32_bf16(false, a, false, b, (short)0, c,
                                                 false, false);
}

// A fragment: 16x32 bf16, row-major source (lda in elements).
// lane<16 : row=lane,    f[0..7]=K0..7,  f[8..15]=K16..23
// lane>=16: row=lane-16, f[0..7]=K8..15, f[8..15]=K24..31
DEVINLINE v16bf a_frag(const __bf16* A, int lda, int lane) {
  int r = lane & 15, h = (lane >> 4) & 1;
  const __bf16* p0 = A + (size_t)r * lda + 8 * h;
  v8bf lo = *(const v8bf*)p0;
  v8bf hi = *(const v8bf*)(p0 + 16);
  v16bf f;
#pragma unroll
  for (int i = 0; i < 8; ++i) { f[i] = lo[i]; f[i + 8] = hi[i]; }
  return f;
}

// A fragment converted on the fly from f32 source.
DEVINLINE v16bf a_frag_f32(const float* A, int lda, int lane) {
  int r = lane & 15, h = (lane >> 4) & 1;
  const float* p0 = A + (size_t)r * lda + 8 * h;
  const float* p1 = p0 + 16;
  v16bf f;
#pragma unroll
  for (int i = 0; i < 8; ++i) { f[i] = (__bf16)p0[i]; f[i + 8] = (__bf16)p1[i]; }
  return f;
}

// B fragment: 32x16 bf16 loaded from B^T stored row-major [N x ldbt].
// lane<16 : col=lane,    f[0..7]=K0..7,  f[8..15]=K16..23
// lane>=16: col=lane-16, f[0..7]=K8..15, f[8..15]=K24..31
DEVINLINE v16bf b_frag(const __bf16* BT, int ldbt, int lane) {
  int cn = lane & 15, h = (lane >> 4) & 1;
  const __bf16* p = BT + (size_t)cn * ldbt + 8 * h;
  v8bf lo = *(const v8bf*)p;
  v8bf hi = *(const v8bf*)(p + 16);
  v16bf f;
#pragma unroll
  for (int i = 0; i < 8; ++i) { f[i] = lo[i]; f[i + 8] = hi[i]; }
  return f;
}

// ---------------------------------------------------------------- prep kernels
__global__ void k_prep_dw(const float* __restrict__ dw, __bf16* __restrict__ dwT) {
  int i = blockIdx.x * blockDim.x + threadIdx.x;
  if (i < DD * DD) {
    int e = i / DD, k = i % DD;
    dwT[(size_t)e * DD + k] = (__bf16)dw[(size_t)k * DD + e];  // dwT[e][k] = dw[k][e]
  }
}

__global__ void k_prep_query(const float* __restrict__ q, const float* __restrict__ qwts,
                             __bf16* __restrict__ qb, __bf16* __restrict__ qbT,
                             float* __restrict__ qw) {
  __shared__ float red[DD];
  int bq = blockIdx.x, b = bq / LQ, lq = bq % LQ;
  int d = threadIdx.x;
  float v = q[((size_t)b * LQ + lq) * DD + d];
  qb[((size_t)b * LQ + lq) * DD + d] = (__bf16)v;
  qbT[((size_t)b * DD + d) * LQ + lq] = (__bf16)v;
  red[d] = v * qwts[d];
  __syncthreads();
#pragma unroll
  for (int s = DD / 2; s > 0; s >>= 1) {
    if (d < s) red[d] += red[d + s];
    __syncthreads();
  }
  if (d == 0) qw[(size_t)b * LQ + lq] = red[0];
}

__global__ void k_prep_context(const float* __restrict__ c, const float* __restrict__ cwts,
                               __bf16* __restrict__ ctxbT, float* __restrict__ cw) {
  __shared__ float red[DD];
  int bc = blockIdx.x, b = bc / LC, lc = bc % LC;
  int d = threadIdx.x;
  float v = c[((size_t)b * LC + lc) * DD + d];
  ctxbT[((size_t)b * DD + d) * LC + lc] = (__bf16)v;
  red[d] = v * cwts[d];
  __syncthreads();
#pragma unroll
  for (int s = DD / 2; s > 0; s >>= 1) {
    if (d < s) red[d] += red[d + s];
    __syncthreads();
  }
  if (d == 0) cw[(size_t)b * LC + lc] = red[0];
}

// --------------------------- WC = ctx @ dotW : one wave = 16x128 strip (8 acc)
__global__ void k_gemm_wc(const float* __restrict__ ctx, const __bf16* __restrict__ dwT,
                          __bf16* __restrict__ wc) {
  int wid = (blockIdx.x * blockDim.x + threadIdx.x) >> 5;
  int lane = threadIdx.x & 31;
  int b = wid / (LC / 16);
  int mt = wid % (LC / 16);
  const float* A = ctx + ((size_t)b * LC + mt * 16) * DD;
  v8f acc[8] = {};
#pragma unroll
  for (int k0 = 0; k0 < DD; k0 += 32) {
    v16bf a = a_frag_f32(A + k0, DD, lane);  // loaded once, used 8x
#pragma unroll
    for (int nt = 0; nt < 8; ++nt)
      acc[nt] = wmma_bf16(a, b_frag(dwT + (size_t)nt * 16 * DD + k0, DD, lane), acc[nt]);
  }
  int cn = lane & 15, rbase = (lane >> 4) * 8;
#pragma unroll
  for (int nt = 0; nt < 8; ++nt)
#pragma unroll
    for (int r = 0; r < 8; ++r)
      wc[((size_t)b * LC + mt * 16 + rbase + r) * DD + nt * 16 + cn] = (__bf16)acc[nt][r];
}

// ------------------------- S = WC @ Q^T + cw + qw + mask, row softmax -> c2q
__global__ void k_sim_rowsoftmax(const __bf16* __restrict__ WC, const __bf16* __restrict__ QB,
                                 const float* __restrict__ CW, const float* __restrict__ QW,
                                 const uint8_t* __restrict__ mask,
                                 __bf16* __restrict__ Sbuf, __bf16* __restrict__ C2Q) {
  __shared__ float St[16][LQ];     // 16 KB
  __shared__ float pmax[16][16];
  __shared__ float psum[16][16];
  __shared__ float rmax[16], rsum[16];

  int b = blockIdx.x / (LC / 16);
  int ct = blockIdx.x % (LC / 16);
  int lane = threadIdx.x & 31;
  int w = threadIdx.x >> 5;  // 8 waves, each does 2 column tiles

  const __bf16* Abase = WC + ((size_t)b * LC + ct * 16) * DD;
  v16bf afr[4];
#pragma unroll
  for (int kk = 0; kk < 4; ++kk) afr[kk] = a_frag(Abase + kk * 32, DD, lane);

#pragma unroll
  for (int sub = 0; sub < 2; ++sub) {
    int nt = w * 2 + sub;
    const __bf16* BTbase = QB + ((size_t)b * LQ + nt * 16) * DD;
    v8f acc = {};
#pragma unroll
    for (int kk = 0; kk < 4; ++kk)
      acc = wmma_bf16(afr[kk], b_frag(BTbase + kk * 32, DD, lane), acc);
    int cn = lane & 15, rbase = (lane >> 4) * 8;
    int col = nt * 16 + cn;
    float qv = QW[(size_t)b * LQ + col];
#pragma unroll
    for (int r = 0; r < 8; ++r) {
      int row = rbase + r;
      int c = ct * 16 + row;
      float cv = CW[(size_t)b * LC + c];
      uint8_t mk = mask[((size_t)b * LC + c) * LQ + col];
      St[row][col] = acc[r] + cv + qv + (mk ? 0.f : -1e9f);
    }
  }
  __syncthreads();

  // dump masked similarity (bf16 -- halves score-matrix traffic)
  size_t sb = ((size_t)b * LC + ct * 16) * LQ;
  const float* flat = &St[0][0];
  for (int i = threadIdx.x; i < 16 * LQ; i += 256) Sbuf[sb + i] = (__bf16)flat[i];

  // row softmax over LQ
  int row = threadIdx.x >> 4, j = threadIdx.x & 15;
  float m = -3.0e38f;
  for (int col = j; col < LQ; col += 16) m = fmaxf(m, St[row][col]);
  pmax[row][j] = m;
  __syncthreads();
  if (j == 0) {
    float mm = pmax[row][0];
#pragma unroll
    for (int i = 1; i < 16; ++i) mm = fmaxf(mm, pmax[row][i]);
    rmax[row] = mm;
  }
  __syncthreads();
  float mm = rmax[row];
  float s = 0.f;
  for (int col = j; col < LQ; col += 16) s += __expf(St[row][col] - mm);
  psum[row][j] = s;
  __syncthreads();
  if (j == 0) {
    float ss = 0.f;
#pragma unroll
    for (int i = 0; i < 16; ++i) ss += psum[row][i];
    rsum[row] = ss;
  }
  __syncthreads();
  for (int i = threadIdx.x; i < 16 * LQ; i += 256) {
    int rr = i >> 8, col = i & 255;
    C2Q[sb + i] = (__bf16)(__expf(St[rr][col] - rmax[rr]) / rsum[rr]);
  }
}

// ----------------------------------------- column (over LC) online softmax stats
__global__ void k_colstats(const __bf16* __restrict__ Sbuf, float* __restrict__ cmax,
                           float* __restrict__ csum) {
  __shared__ float sm[4][64], ss[4][64];
  int b = blockIdx.x / (LQ / 64);
  int qt = blockIdx.x % (LQ / 64);
  int t = threadIdx.x;
  int q = qt * 64 + (t & 63);
  int g = t >> 6;  // 4 row-groups
  float m = -3.0e38f, s = 0.f;
  const __bf16* base = Sbuf + (size_t)b * LC * LQ + q;
  for (int c = g; c < LC; c += 4) {
    const __bf16* p = base + (size_t)c * LQ;
    __builtin_prefetch(p + (size_t)64 * LQ, 0, 1);  // global_prefetch_b8, 64 rows ahead
    float x = (float)*p;
    if (x > m) { s *= __expf(m - x); m = x; }
    s += __expf(x - m);
  }
  sm[g][t & 63] = m;
  ss[g][t & 63] = s;
  __syncthreads();
  if (g == 0) {
    float M = m, S = s;
#pragma unroll
    for (int i = 1; i < 4; ++i) {
      float m2 = sm[i][t & 63], s2 = ss[i][t & 63];
      float nM = fmaxf(M, m2);
      S = S * __expf(M - nM) + s2 * __expf(m2 - nM);
      M = nM;
    }
    cmax[(size_t)b * LQ + q] = M;
    csum[(size_t)b * LQ + q] = S;
  }
}

// ------------------------------- q2c^T (bf16, [B, LQ, LC]) via LDS tile transpose
__global__ void k_q2ct(const __bf16* __restrict__ Sbuf, const float* __restrict__ cmax,
                       const float* __restrict__ csum, __bf16* __restrict__ q2cT) {
  __shared__ float tile[64][65];
  int b = blockIdx.z, ct = blockIdx.y, qt = blockIdx.x;
  for (int i = threadIdx.x; i < 64 * 64; i += 256) {
    int r = i >> 6, col = i & 63;  // r: c offset, col: q offset
    tile[r][col] = (float)Sbuf[((size_t)b * LC + ct * 64 + r) * LQ + qt * 64 + col];
  }
  __syncthreads();
  for (int i = threadIdx.x; i < 64 * 64; i += 256) {
    int qr = i >> 6, cc = i & 63;
    int q = qt * 64 + qr, c = ct * 64 + cc;
    float M = cmax[(size_t)b * LQ + q], S = csum[(size_t)b * LQ + q];
    q2cT[((size_t)b * LQ + q) * LC + c] = (__bf16)(__expf(tile[cc][qr] - M) / S);
  }
}

// ------------- S2C^T[b,d,q] = (q2c^T @ context)^T : one wave = 16x64 (4 acc)
__global__ void k_gemm_s2c(const __bf16* __restrict__ q2cT, const __bf16* __restrict__ ctxbT,
                           __bf16* __restrict__ s2cT) {
  int wid = (blockIdx.x * blockDim.x + threadIdx.x) >> 5;
  int lane = threadIdx.x & 31;
  constexpr int MT = LQ / 16;  // 16
  int b = wid / (MT * 2);
  int rem = wid % (MT * 2);
  int mt = rem / 2, ng = rem % 2;  // ng: which 64-wide feature group
  const __bf16* A = q2cT + ((size_t)b * LQ + mt * 16) * LC;                // [LQ x LC]
  const __bf16* BT = ctxbT + ((size_t)b * DD + ng * 64) * LC;              // ctx^T
  v8f acc[4] = {};
  for (int k0 = 0; k0 < LC; k0 += 32) {
    v16bf a = a_frag(A + k0, LC, lane);  // loaded once, used 4x
#pragma unroll
    for (int j = 0; j < 4; ++j)
      acc[j] = wmma_bf16(a, b_frag(BT + (size_t)j * 16 * LC + k0, LC, lane), acc[j]);
  }
  int cn = lane & 15, rbase = (lane >> 4) * 8;
  int qbase = mt * 16 + rbase;
#pragma unroll
  for (int j = 0; j < 4; ++j) {
    int d = ng * 64 + j * 16 + cn;
    v8bf outv;
#pragma unroll
    for (int r = 0; r < 8; ++r) outv[r] = (__bf16)acc[j][r];
    *(v8bf*)(s2cT + ((size_t)b * DD + d) * LQ + qbase) = outv;  // transposed store
  }
}

// ------------- ctq = c2q @ Q ; qtc = c2q @ S2C ; assemble [ctx, ctq, ctx*ctq, ctx*qtc]
__global__ void k_final(const __bf16* __restrict__ C2Q, const __bf16* __restrict__ qbT,
                        const __bf16* __restrict__ s2cT, const float* __restrict__ ctx,
                        float* __restrict__ out) {
  int b = blockIdx.x / (LC / 16);
  int ct = blockIdx.x % (LC / 16);
  int lane = threadIdx.x & 31;
  int nt = threadIdx.x >> 5;  // 8 waves -> 8 feature tiles of 16

  const __bf16* A = C2Q + ((size_t)b * LC + ct * 16) * LQ;     // [16 x 256]
  const __bf16* BT1 = qbT + ((size_t)b * DD + nt * 16) * LQ;   // Q^T   [D x LQ]
  const __bf16* BT2 = s2cT + ((size_t)b * DD + nt * 16) * LQ;  // S2C^T [D x LQ]
  v8f acc1 = {}, acc2 = {};
#pragma unroll
  for (int k0 = 0; k0 < LQ; k0 += 32) {
    v16bf a = a_frag(A + k0, LQ, lane);  // shared A fragment for both GEMMs
    acc1 = wmma_bf16(a, b_frag(BT1 + k0, LQ, lane), acc1);
    acc2 = wmma_bf16(a, b_frag(BT2 + k0, LQ, lane), acc2);
  }
  int cn = lane & 15, rbase = (lane >> 4) * 8;
  int d = nt * 16 + cn;
#pragma unroll
  for (int r = 0; r < 8; ++r) {
    int c = ct * 16 + rbase + r;
    float cx = ctx[((size_t)b * LC + c) * DD + d];
    float* o = out + ((size_t)b * LC + c) * (4 * DD);
    o[d] = cx;
    o[DD + d] = acc1[r];
    o[2 * DD + d] = cx * acc1[r];
    o[3 * DD + d] = cx * acc2[r];
  }
}

// ---------------------------------------------------------------- launch
extern "C" void kernel_launch(void* const* d_in, const int* in_sizes, int n_in,
                              void* d_out, int out_size, void* d_ws, size_t ws_size,
                              hipStream_t stream) {
  const float* query = (const float*)d_in[0];     // [B, LQ, D]
  const float* ctx   = (const float*)d_in[1];     // [B, LC, D]
  const uint8_t* mask = (const uint8_t*)d_in[2];  // [B, LC, LQ] bool
  const float* qwts  = (const float*)d_in[3];     // [D, 1]
  const float* cwts  = (const float*)d_in[4];     // [D, 1]
  const float* dw    = (const float*)d_in[5];     // [D, D]
  float* out = (float*)d_out;                     // [B, LC, 4D]

  char* ws = (char*)d_ws;
  size_t off = 0;
  auto take = [&](size_t bytes) {
    void* p = ws + off;
    off = (off + bytes + 255) & ~(size_t)255;
    return p;
  };
  __bf16* Sbuf  = (__bf16*)take((size_t)NB * LC * LQ * 2);  // masked similarity (bf16)
  __bf16* c2q   = (__bf16*)take((size_t)NB * LC * LQ * 2);  // row softmax
  __bf16* q2cT  = (__bf16*)take((size_t)NB * LQ * LC * 2);  // col softmax, transposed
  __bf16* wc    = (__bf16*)take((size_t)NB * LC * DD * 2);  // ctx @ dotW
  __bf16* ctxbT = (__bf16*)take((size_t)NB * DD * LC * 2);  // ctx^T bf16
  __bf16* qb    = (__bf16*)take((size_t)NB * LQ * DD * 2);  // Q bf16
  __bf16* qbT   = (__bf16*)take((size_t)NB * DD * LQ * 2);  // Q^T bf16
  __bf16* s2cT  = (__bf16*)take((size_t)NB * DD * LQ * 2);  // (q2c^T @ ctx)^T
  __bf16* dwT   = (__bf16*)take((size_t)DD * DD * 2);       // dotW^T bf16
  float*  cw    = (float*)take((size_t)NB * LC * 4);
  float*  qw    = (float*)take((size_t)NB * LQ * 4);
  float*  cmax  = (float*)take((size_t)NB * LQ * 4);
  float*  csum  = (float*)take((size_t)NB * LQ * 4);

  k_prep_dw<<<(DD * DD + 255) / 256, 256, 0, stream>>>(dw, dwT);
  k_prep_query<<<NB * LQ, DD, 0, stream>>>(query, qwts, qb, qbT, qw);
  k_prep_context<<<NB * LC, DD, 0, stream>>>(ctx, cwts, ctxbT, cw);

  // WC: one wave per 16-row strip -> 8192 waves / 8 waves per block
  k_gemm_wc<<<NB * (LC / 16) / 8, 256, 0, stream>>>(ctx, dwT, wc);

  // similarity + mask + row softmax (one block = 16 context rows)
  k_sim_rowsoftmax<<<NB * (LC / 16), 256, 0, stream>>>(wc, qb, cw, qw, mask, Sbuf, c2q);

  k_colstats<<<NB * (LQ / 64), 256, 0, stream>>>(Sbuf, cmax, csum);
  k_q2ct<<<dim3(LQ / 64, LC / 64, NB), 256, 0, stream>>>(Sbuf, cmax, csum, q2cT);

  // S2C: one wave per 16x64 tile -> 2048 waves / 8 waves per block
  k_gemm_s2c<<<NB * (LQ / 16) * (DD / 64) / 8, 256, 0, stream>>>(q2cT, ctxbT, s2cT);

  k_final<<<NB * (LC / 16), 256, 0, stream>>>(c2q, qbT, s2cT, ctx, out);
}
